// QKPosVTokAttention_40046275068145
// MI455X (gfx1250) — matrix operand order repaired and checked by reading, not compiled, not run on hardware
//
#include <hip/hip_runtime.h>
#include <hip/hip_bf16.h>

// ---------------------------------------------------------------------------
// Causal MHA: q=x_pos@Wq (scaled), k=x_pos@Wk, v=x_tok@Wv, flash attention,
// out = attn_out @ Wo.   B=2, T=4096, H=8, HD=32, D_MODEL=256, x dim 128.
// All GEMM work on v_wmma_f32_16x16x32_f16 (wave32, gfx1250).
//  - Weights pre-converted to f16 transposed [N][K]: every WMMA B operand is
//    two 16-byte loads per lane.
//  - Flash attention: unmasked steady-state loop + masked diagonal tail;
//    row-sums via a ones-matrix WMMA; row-max via packed-f16 DPP butterfly
//    (v_pk_max_num_f16 + dpp16) -> no ds_bpermute in the hot loop.
//  - WGP-scope prefetch (inline asm, default scope 0) of the next K/V block:
//    fills the near cache; SE/DEV-scope prefetch only touches GL2, which is
//    already resident (working set ~20 MB of 192 MB L2).
// ---------------------------------------------------------------------------

typedef _Float16 h16;
typedef __attribute__((ext_vector_type(16))) _Float16 v16h;
typedef __attribute__((ext_vector_type(8)))  _Float16 v8h;
typedef __attribute__((ext_vector_type(2)))  _Float16 v2h;
typedef __attribute__((ext_vector_type(8)))  float    v8f;

#define TT 4096
#define NB 2
#define NH 8
#define HD 32
#define DM 256
#define XD 128
#define NEG_BIG (-3.0e38f)

// DPP16 controls (operate within 16-lane rows; wave32 = 2 rows).
#define DPP_QUAD_XOR1   0xB1   // quad_perm [1,0,3,2]
#define DPP_QUAD_XOR2   0x4E   // quad_perm [2,3,0,1]
#define DPP_HALF_MIRROR 0x141  // lane i <-> 7-i within each 8
#define DPP_MIRROR      0x140  // lane i <-> 15-i within each 16

static __device__ __forceinline__ v16h cat8(v8h lo, v8h hi) {
  return __builtin_shufflevector(lo, hi, 0,1,2,3,4,5,6,7,8,9,10,11,12,13,14,15);
}

static __device__ __forceinline__ v8f wmma_f16(v16h a, v16h b, v8f c) {
  // D = A(16x32 f16) * B(32x16 f16) + C(16x16 f32)
  return __builtin_amdgcn_wmma_f32_16x16x32_f16(
      /*neg_a=*/false, a, /*neg_b=*/false, b,
      /*c_mod=*/(short)0, c, /*reuse_a=*/false, /*reuse_b=*/false);
}

template <int CTRL>
static __device__ __forceinline__ v2h dpp_pkmax(v2h v) {
  const unsigned u = __builtin_bit_cast(unsigned, v);
  const unsigned s = (unsigned)__builtin_amdgcn_update_dpp(
      (int)u, (int)u, CTRL, 0xf, 0xf, true);
  return __builtin_elementwise_max(v, __builtin_bit_cast(v2h, s));
}

// WGP-scope prefetch: ISA default scope 0 pulls into all cache levels on a
// miss (the builtin only reaches SE/DEV scope, which skip the WGP cache).
static __device__ __forceinline__ void prefetch_wgp(const void* p) {
  asm volatile("global_prefetch_b8 %0, off" :: "v"(p));
}

// A-matrix (16x32 f16) half-index for lane-group g, vgpr v, pos p:
//   K = (v<4 ? 0 : 16) + g*8 + (v&3)*2 + p     (ISA 7.12.2)
// B-matrix (32x16 f16):
//   K = same as A ; N = lane&15           -> two 16B chunks per lane
// C/D (16x16 f32): M = r + 8*g ; N = lane&15

// ---------------------------------------------------------------------------
// Kernel 0: weight convert+transpose, f32 [K][N] -> f16 [N][K].
// ---------------------------------------------------------------------------
__global__ __launch_bounds__(256) void wtrans_kernel(
    const float* __restrict__ W, h16* __restrict__ WT, int K, int N)
{
  const int i = blockIdx.x * blockDim.x + threadIdx.x;   // over N*K
  if (i < N * K) {
    const int n = i / K, k = i - n * K;
    WT[i] = (h16)W[(size_t)k * N + n];
  }
}

// ---------------------------------------------------------------------------
// Kernel 1: QKV projection (templated on MODE: 0=Q, 1=K, 2=V).
// grid.x = 8192/16 row tiles; 8 waves/block, wave w does column tiles {w,w+8}.
// ---------------------------------------------------------------------------
template <int MODE>
__global__ __launch_bounds__(256) void qkv_proj_kernel(
    const float* __restrict__ x, const h16* __restrict__ WT,  // [256][64] f16
    h16* __restrict__ dst)
{
  const int lane = threadIdx.x & 31;
  const int wave = threadIdx.x >> 5;
  const int g    = lane >> 4;
  const int nl   = lane & 15;
  const int m0   = blockIdx.x * 16;       // global row tile (B*T flattened)
  const int xoff = (MODE == 2) ? 0 : 64;  // V uses x_tok, Q/K use x_pos

  // A operands for the two K-steps (K=64 -> 2 x 32), converted f32->f16.
  const float* xr = x + (size_t)(m0 + nl) * XD + xoff;
  v16h a0, a1;
  #pragma unroll
  for (int v = 0; v < 8; ++v)
    #pragma unroll
    for (int p = 0; p < 2; ++p) {
      const int K = ((v < 4) ? 0 : 16) + g * 8 + (v & 3) * 2 + p;
      a0[v * 2 + p] = (h16)xr[K];
      a1[v * 2 + p] = (h16)xr[32 + K];
    }

  #pragma unroll
  for (int it = 0; it < 2; ++it) {
    const int nt = wave + it * 8;
    const int c  = nt * 16 + nl;          // output channel 0..255
    const h16* wr = WT + (size_t)c * 64;   // [N][K=64] f16
    v8f acc = {};
    acc = wmma_f16(a0, cat8(*(const v8h*)(wr + g * 8), *(const v8h*)(wr + 16 + g * 8)), acc);
    acc = wmma_f16(a1, cat8(*(const v8h*)(wr + 32 + g * 8), *(const v8h*)(wr + 48 + g * 8)), acc);

    const int hh = c >> 5, d = c & 31;    // head, head-dim
    #pragma unroll
    for (int r = 0; r < 8; ++r) {
      const int m  = m0 + r + 8 * g;      // flattened token
      const int bb = m >> 12;             // batch (T = 4096)
      const int t  = m & (TT - 1);
      const size_t bh = (size_t)bb * NH + hh;
      if (MODE == 0)
        dst[(bh * TT + t) * HD + d] = (h16)(acc[r] * 0.17677669529663687f);
      else if (MODE == 1)
        dst[(bh * TT + t) * HD + d] = (h16)acc[r];
      else
        dst[(bh * HD + d) * TT + t] = (h16)acc[r];  // V transposed [dim][t]
    }
  }
}

// ---------------------------------------------------------------------------
// One 32-key flash-attention block step (per wave, 16 query rows).
// MASKED selects the causal-masked diagonal path; the steady-state path has
// no compares.  5 WMMAs per block: 2 score, 1 row-sum (ones), 2 P*V.
// ---------------------------------------------------------------------------
template <bool MASKED>
static __device__ __forceinline__ void attn_block(
    int kbase, int q0, int g, int nl,
    const h16* __restrict__ Kb, const h16* __restrict__ Vb,
    const v16h& aq, const v16h& ones, _Float16 (*Pl)[32],
    v8f& acc0, v8f& acc1, float mr[8], float lr[8])
{
  // K tile as B operand: n = key (nl / nl+16), K-dim = head dims (contig).
  const h16* kr0 = Kb + (size_t)(kbase + nl) * HD;
  const v16h bk0 = cat8(*(const v8h*)(kr0 + g * 8), *(const v8h*)(kr0 + 16 + g * 8));
  const v16h bk1 = cat8(*(const v8h*)(kr0 + 16 * HD + g * 8), *(const v8h*)(kr0 + 16 * HD + 16 + g * 8));

  const v8f z = {};
  v8f s0 = wmma_f16(aq, bk0, z);       // 16q x 16k, keys kbase..+15
  v8f s1 = wmma_f16(aq, bk1, z);       // keys kbase+16..+31

  // Per-lane row candidate max (and causal mask on the diagonal tail).
  const int key0 = kbase + nl, key1 = key0 + 16;
  float mb[8];
  #pragma unroll
  for (int r = 0; r < 8; ++r) {
    float v0 = s0[r], v1 = s1[r];
    if (MASKED) {
      const int qa = q0 + r + 8 * g;
      v0 = (key0 <= qa) ? v0 : NEG_BIG;
      v1 = (key1 <= qa) ? v1 : NEG_BIG;
      s0[r] = v0; s1[r] = v1;
    }
    mb[r] = fmaxf(v0, v1);
  }

  // Row max across the 16-lane half: rows packed pairwise into f16 and
  // reduced with a 4-step DPP butterfly (VALU only, no DS traffic).
  // f16-rounded max is a valid softmax shift (applied consistently below).
  v2h pk[4];
  #pragma unroll
  for (int j = 0; j < 4; ++j) {
    v2h t; t[0] = (h16)mb[2 * j]; t[1] = (h16)mb[2 * j + 1];
    t = dpp_pkmax<DPP_QUAD_XOR1>(t);
    t = dpp_pkmax<DPP_QUAD_XOR2>(t);
    t = dpp_pkmax<DPP_HALF_MIRROR>(t);
    t = dpp_pkmax<DPP_MIRROR>(t);
    pk[j] = t;
  }

  // exp + accumulator rescale.
  float fsc[8], p0[8], p1[8];
  #pragma unroll
  for (int r = 0; r < 8; ++r) {
    const float mn = fmaxf(mr[r], (float)pk[r >> 1][r & 1]);
    fsc[r] = __expf(mr[r] - mn);
    p0[r]  = __expf(s0[r] - mn);
    p1[r]  = __expf(s1[r] - mn);
    mr[r]  = mn;
    acc0[r] *= fsc[r];
    acc1[r] *= fsc[r];
  }

  // D-layout -> A-layout via per-wave LDS bounce (same-wave DS is in-order).
  #pragma unroll
  for (int r = 0; r < 8; ++r) {
    Pl[r + 8 * g][nl]      = (h16)p0[r];
    Pl[r + 8 * g][nl + 16] = (h16)p1[r];
  }
  asm volatile("s_wait_dscnt 0" ::: "memory");
  const _Float16* pr = &Pl[nl][0];
  const v16h ap = cat8(*(const v8h*)(pr + g * 8),
                       *(const v8h*)(pr + 16 + g * 8));
  asm volatile("" ::: "memory");       // keep next-iter stores after these loads

  // Row-sum of P via ones-matrix WMMA; uses the same f16 P that multiplies V.
  const v8f lb = wmma_f16(ap, ones, z);
  #pragma unroll
  for (int r = 0; r < 8; ++r) lr[r] = lr[r] * fsc[r] + lb[r];

  // V^T tile as B operand: n = head-dim, K-dim = keys (contiguous).
  const h16* vr0 = Vb + (size_t)nl * TT + kbase;
  const h16* vr1 = Vb + (size_t)(nl + 16) * TT + kbase;
  acc0 = wmma_f16(ap, cat8(*(const v8h*)(vr0 + g * 8), *(const v8h*)(vr0 + 16 + g * 8)), acc0);
  acc1 = wmma_f16(ap, cat8(*(const v8h*)(vr1 + g * 8), *(const v8h*)(vr1 + 16 + g * 8)), acc1);
}

// ---------------------------------------------------------------------------
// Kernel 2: flash attention.  grid = (T/64, H, B); 4 waves/block, each wave
// owns one 16-row query tile; loop over 32-key blocks with online softmax.
// ---------------------------------------------------------------------------
__global__ __launch_bounds__(128) void flash_attn_kernel(
    const h16* __restrict__ qs, const h16* __restrict__ ksm,
    const h16* __restrict__ vt, h16* __restrict__ ao)
{
  __shared__ _Float16 Pl[4][16][32];   // per-wave P bounce (1 KB each)

  const int lane = threadIdx.x & 31;
  const int wave = threadIdx.x >> 5;
  const int g    = lane >> 4;
  const int nl   = lane & 15;
  const int hh   = blockIdx.y;
  const int bb   = blockIdx.z;
  const int q0   = blockIdx.x * 64 + wave * 16;

  const size_t bh = (size_t)bb * NH + hh;
  const h16* Qb = qs  + bh * (size_t)TT * HD;
  const h16* Kb = ksm + bh * (size_t)TT * HD;
  const h16* Vb = vt  + bh * (size_t)HD * TT;

  // Q tile as A operand (pre-scaled). Row = q0 + nl; two 16B chunks.
  const h16* qr = Qb + (size_t)(q0 + nl) * HD;
  const v16h aq = cat8(*(const v8h*)(qr + g * 8),
                       *(const v8h*)(qr + 16 + g * 8));

  v16h ones;
  #pragma unroll
  for (int i = 0; i < 16; ++i) ones[i] = (h16)1.0f;

  v8f acc0 = {}, acc1 = {};            // O accumulator: dims 0-15 / 16-31
  float mr[8], lr[8];
  #pragma unroll
  for (int r = 0; r < 8; ++r) { mr[r] = NEG_BIG; lr[r] = 0.0f; }

  const int nfull = q0 >> 5;           // blocks with kbase+31 <= q0: no mask
  const int nkb   = (q0 + 47) >> 5;    // all blocks covering keys <= q0+15

  int kb = 0;
  for (; kb < nfull; ++kb) {
    const int kbase = kb * 32;
    // WGP-scope prefetch of the next K/V block (fills the near cache).
    prefetch_wgp(Kb + (size_t)(kbase + 32 + nl) * HD + g * 16);
    prefetch_wgp(Vb + (size_t)nl * TT + kbase + 32);
    attn_block<false>(kbase, q0, g, nl, Kb, Vb, aq, ones, Pl[wave],
                      acc0, acc1, mr, lr);
  }
  for (; kb < nkb; ++kb) {
    attn_block<true>(kb * 32, q0, g, nl, Kb, Vb, aq, ones, Pl[wave],
                     acc0, acc1, mr, lr);
  }

  // Epilogue: normalize and store f16 attention output [B*T, 256].
  #pragma unroll
  for (int r = 0; r < 8; ++r) {
    const int t = q0 + r + 8 * g;
    const size_t rowb = ((size_t)bb * TT + t) * DM + hh * HD;
    const float inv = 1.0f / lr[r];
    ao[rowb + nl]      = (h16)(acc0[r] * inv);
    ao[rowb + nl + 16] = (h16)(acc1[r] * inv);
  }
}

// ---------------------------------------------------------------------------
// Kernel 3: out = attn_out(f16) @ Wo^T(f16 [256][256]), f32 result.
// ---------------------------------------------------------------------------
__global__ __launch_bounds__(256) void out_proj_kernel(
    const h16* __restrict__ ao, const h16* __restrict__ WoT,
    float* __restrict__ out)
{
  const int lane = threadIdx.x & 31;
  const int wave = threadIdx.x >> 5;
  const int g    = lane >> 4;
  const int nl   = lane & 15;
  const int m0   = blockIdx.x * 16;

  const h16* ar = ao + (size_t)(m0 + nl) * DM;

  #pragma unroll
  for (int it = 0; it < 2; ++it) {
    const int nt = wave + it * 8;
    const int n  = nt * 16 + nl;
    const h16* wr = WoT + (size_t)n * DM;   // [N][K=256] f16
    v8f acc = {};
    #pragma unroll
    for (int ks0 = 0; ks0 < DM; ks0 += 32) {
      const v16h a = cat8(*(const v8h*)(ar + ks0 + g * 8),
                          *(const v8h*)(ar + ks0 + 16 + g * 8));
      const v16h b = cat8(*(const v8h*)(wr + ks0 + g * 8),
                          *(const v8h*)(wr + ks0 + 16 + g * 8));
      acc = wmma_f16(a, b, acc);
    }
    #pragma unroll
    for (int r = 0; r < 8; ++r)
      out[(size_t)(m0 + r + 8 * g) * DM + n] = acc[r];
  }
}

// ---------------------------------------------------------------------------
extern "C" void kernel_launch(void* const* d_in, const int* in_sizes, int n_in,
                              void* d_out, int out_size, void* d_ws, size_t ws_size,
                              hipStream_t stream) {
  const float* x  = (const float*)d_in[0];   // [2,4096,128]
  const float* Wq = (const float*)d_in[1];   // [64,256]
  const float* Wk = (const float*)d_in[2];   // [64,256]
  const float* Wv = (const float*)d_in[3];   // [64,256]
  const float* Wo = (const float*)d_in[4];   // [256,256]
  float* out = (float*)d_out;                // [2,4096,256] f32

  const size_t NTOK = (size_t)NB * TT;             // 8192
  const size_t HTD  = (size_t)NB * NH * TT * HD;   // 2,097,152 halves (4 MB)

  h16* qs  = (h16*)d_ws;          // [B,H,T,32] pre-scaled       4 MB
  h16* ksm = qs  + HTD;           // [B,H,T,32]                  4 MB
  h16* vt  = ksm + HTD;           // [B,H,32,T] transposed       4 MB
  h16* ao  = vt  + HTD;           // [B*T,256]                   4 MB
  h16* wqT = ao  + NTOK * DM;     // [256][64]                   32 KB
  h16* wkT = wqT + (size_t)DM * 64;
  h16* wvT = wkT + (size_t)DM * 64;
  h16* woT = wvT + (size_t)DM * 64;  // [256][256]              128 KB

  // Weight convert + transpose (tiny, L2-resident afterwards).
  wtrans_kernel<<<dim3((DM * 64 + 255) / 256), 256, 0, stream>>>(Wq, wqT, 64, DM);
  wtrans_kernel<<<dim3((DM * 64 + 255) / 256), 256, 0, stream>>>(Wk, wkT, 64, DM);
  wtrans_kernel<<<dim3((DM * 64 + 255) / 256), 256, 0, stream>>>(Wv, wvT, 64, DM);
  wtrans_kernel<<<dim3((DM * DM + 255) / 256), 256, 0, stream>>>(Wo, woT, DM, DM);

  qkv_proj_kernel<0><<<dim3(NTOK / 16), 256, 0, stream>>>(x, wqT, qs);
  qkv_proj_kernel<1><<<dim3(NTOK / 16), 256, 0, stream>>>(x, wkT, ksm);
  qkv_proj_kernel<2><<<dim3(NTOK / 16), 256, 0, stream>>>(x, wvT, vt);

  flash_attn_kernel<<<dim3(TT / 64, NH, NB), 128, 0, stream>>>(qs, ksm, vt, ao);

  out_proj_kernel<<<dim3(NTOK / 16), 256, 0, stream>>>(ao, woT, out);
}
